// MultiHeadAttention_56616258896415
// MI455X (gfx1250) — matrix operand order repaired and checked by reading, not compiled, not run on hardware
//
#include <hip/hip_runtime.h>

typedef _Float16 half_t;
typedef __attribute__((ext_vector_type(16))) _Float16 v16h;
typedef __attribute__((ext_vector_type(8)))  _Float16 v8h;
typedef __attribute__((ext_vector_type(8)))  float    v8f;

#define B_  4
#define S_  2048
#define E_  1024
#define H_  16
#define D_  64
#define HD_ (H_ * D_)

// ---------------- fragment helpers (CDNA5 WMMA 16x16x32 f16 layouts) -------

static __device__ __forceinline__ v8f vzero8() {
  v8f v;
#pragma unroll
  for (int i = 0; i < 8; ++i) v[i] = 0.0f;
  return v;
}

static __device__ __forceinline__ v8f wmma_f16(v16h a, v16h b, v8f c) {
  return __builtin_amdgcn_wmma_f32_16x16x32_f16(
      /*neg_a=*/false, a, /*neg_b=*/false, b,
      /*c_mod=*/(short)0, c, /*reuse_a=*/false, /*reuse_b=*/false);
}

// A-matrix 16x32 f16 fragment. p -> A[row0][k0], row stride = ld halves.
// lane m = L%16 holds row m; element e: K = g*8+e (e<8), 16+g*8+(e-8) (e>=8).
static __device__ __forceinline__ v16h ld_a16x32(const half_t* p, int ld,
                                                 int m, int g) {
  const half_t* r = p + (size_t)m * ld + g * 8;
  v8h lo = *(const v8h*)(r);
  v8h hi = *(const v8h*)(r + 16);
  v16h a;
#pragma unroll
  for (int i = 0; i < 8; ++i) { a[i] = lo[i]; a[8 + i] = hi[i]; }
  return a;
}

// B-matrix 32x16 f16 fragment, lane (n,g): element e: K = g*8+e (e<8),
// 16+g*8+(e-8) (e>=8); B[K][n] located at p + n*strideN + K.
static __device__ __forceinline__ v16h ld_b32x16(const half_t* p, int strideN,
                                                 int nl, int g) {
  const half_t* r = p + (size_t)nl * strideN + g * 8;
  v8h lo = *(const v8h*)(r);
  v8h hi = *(const v8h*)(r + 16);
  v16h b;
#pragma unroll
  for (int i = 0; i < 8; ++i) { b[i] = lo[i]; b[8 + i] = hi[i]; }
  return b;
}

static __device__ __forceinline__ float rmax16(float v) {
#pragma unroll
  for (int m = 1; m < 16; m <<= 1) v = fmaxf(v, __shfl_xor(v, m, 32));
  return v;
}
static __device__ __forceinline__ float rsum16(float v) {
#pragma unroll
  for (int m = 1; m < 16; m <<= 1) v += __shfl_xor(v, m, 32);
  return v;
}

// gfx1250 async global->LDS copy (ASYNCcnt-tracked DMA path).
// lds_off = group-segment byte offset (low 32 bits of flat shared address).
static __device__ __forceinline__ void async_b128(unsigned lds_off,
                                                  const void* gaddr) {
  asm volatile("global_load_async_to_lds_b128 %0, %1, off"
               :: "v"(lds_off), "v"(gaddr)
               : "memory");
}
static __device__ __forceinline__ unsigned lds_offset(const void* p) {
  return (unsigned)(uintptr_t)p;  // flat LDS aperture: addr[31:0] == LDS offset
}

// ---------------- prep kernels ---------------------------------------------

__global__ void cvt_f32_to_f16(const float* __restrict__ in,
                               half_t* __restrict__ out, int n) {
  int i = blockIdx.x * blockDim.x + threadIdx.x;
  if (i < n) out[i] = (half_t)in[i];
}

// in [Hh, R, C] f32 -> out [Hh, C, R] f16  (per-slice transpose)
__global__ void transpose_w(const float* __restrict__ in,
                            half_t* __restrict__ out, int Hh, int R, int C) {
  long i = (long)blockIdx.x * blockDim.x + threadIdx.x;
  long total = (long)Hh * R * C;
  if (i < total) {
    int c = (int)(i % C);
    long t = i / C;
    int r = (int)(t % R);
    int hh = (int)(t / R);
    out[((long)hh * C + c) * R + r] = (half_t)in[i];
  }
}

// ---------------- fused QKV projection -------------------------------------
// x_h [B*S, E] f16 row-major; W*T [H, D, E] f16 (transposed weights).
// q,k -> [B,H,S,D] f16 ; v -> [B,H,D,S] f16 (transposed for P*V B-frags).
__global__ __launch_bounds__(256)
void qkv_gemm(const half_t* __restrict__ xh,
              const half_t* __restrict__ wqT, const half_t* __restrict__ wkT,
              const half_t* __restrict__ wvT,
              const float* __restrict__ bq, const float* __restrict__ bk,
              const float* __restrict__ bv,
              half_t* __restrict__ q, half_t* __restrict__ k,
              half_t* __restrict__ vt) {
  const int lane = threadIdx.x & 31;
  const int wave = threadIdx.x >> 5;
  const int nl = lane & 15, g = lane >> 4;
  const int h = blockIdx.y;
  const int m0 = blockIdx.x * 128 + wave * 16;  // row in [B*S]

  v8f aq[4], ak[4], av[4];
#pragma unroll
  for (int n = 0; n < 4; ++n) { aq[n] = vzero8(); ak[n] = vzero8(); av[n] = vzero8(); }

  const half_t* xa = xh + (size_t)m0 * E_;
  const half_t* wq0 = wqT + (size_t)h * D_ * E_;
  const half_t* wk0 = wkT + (size_t)h * D_ * E_;
  const half_t* wv0 = wvT + (size_t)h * D_ * E_;

  for (int kc = 0; kc < E_ / 32; ++kc) {
    v16h a = ld_a16x32(xa + kc * 32, E_, nl, g);
#pragma unroll
    for (int n = 0; n < 4; ++n) {
      v16h b;
      b = ld_b32x16(wq0 + (size_t)(n * 16) * E_ + kc * 32, E_, nl, g);
      aq[n] = wmma_f16(a, b, aq[n]);
      b = ld_b32x16(wk0 + (size_t)(n * 16) * E_ + kc * 32, E_, nl, g);
      ak[n] = wmma_f16(a, b, ak[n]);
      b = ld_b32x16(wv0 + (size_t)(n * 16) * E_ + kc * 32, E_, nl, g);
      av[n] = wmma_f16(a, b, av[n]);
    }
  }

  const int b_ = m0 / S_;
  const int s0 = m0 % S_;
  const size_t bh = (size_t)b_ * H_ + h;
#pragma unroll
  for (int n = 0; n < 4; ++n) {
    const int d = n * 16 + nl;
    const float biq = bq[h * D_ + d], bik = bk[h * D_ + d], biv = bv[h * D_ + d];
#pragma unroll
    for (int i = 0; i < 8; ++i) {
      const int srow = s0 + g * 8 + i;
      q[(bh * S_ + srow) * D_ + d] = (half_t)(aq[n][i] + biq);
      k[(bh * S_ + srow) * D_ + d] = (half_t)(ak[n][i] + bik);
      vt[(bh * D_ + d) * S_ + srow] = (half_t)(av[n][i] + biv);
    }
  }
}

// ---------------- flash attention ------------------------------------------
// q,k [B,H,S,D] f16 ; vt [B,H,D,S] f16 ; out attn [B,S,H*D] f16.
// Block = 8 waves covering 128 q rows of one (b,h). K/V tiles (32 kv cols)
// are staged once per block into LDS with async b128 DMA, double-buffered.
__global__ __launch_bounds__(256)
void flash_attn(const half_t* __restrict__ q, const half_t* __restrict__ k,
                const half_t* __restrict__ vt, half_t* __restrict__ attn) {
  const int tid = threadIdx.x;
  const int lane = tid & 31;
  const int wave = tid >> 5;
  const int nl = lane & 15, g = lane >> 4;
  const int bh = blockIdx.y;
  const int q0 = blockIdx.x * 128;
  const int qbase = q0 + wave * 16;

  __shared__ __align__(16) half_t kbuf[2][32 * 64];  // [kv][d]   4KB each
  __shared__ __align__(16) half_t vbuf[2][64 * 32];  // [d][kv]   4KB each
  __shared__ __align__(16) half_t plds[8][16 * 32];  // per-wave P transpose
  half_t* myp = plds[wave];

  const half_t* qp = q + ((size_t)bh * S_ + qbase) * D_;
  const half_t* kp = k + (size_t)bh * S_ * D_;
  const half_t* vp = vt + (size_t)bh * D_ * S_;

  const v16h qa0 = ld_a16x32(qp, D_, nl, g);
  const v16h qa1 = ld_a16x32(qp + 32, D_, nl, g);

  float m_i[8], l_i[8];
  v8f acc[4];
#pragma unroll
  for (int i = 0; i < 8; ++i) { m_i[i] = -3.0e38f; l_i[i] = 0.0f; }
#pragma unroll
  for (int d = 0; d < 4; ++d) acc[d] = vzero8();

  // staging lambda: one async b128 per thread for K, one for V (4KB each)
  const int krow = tid >> 3, kc8 = tid & 7;   // K tile: 32 rows x 128B
  const int vrow = tid >> 2, vc8 = tid & 3;   // V tile: 64 rows x 64B
  auto stage = [&](int buf, int j) {
    async_b128(lds_offset(&kbuf[buf][0]) + (unsigned)tid * 16,
               kp + (size_t)(j + krow) * D_ + kc8 * 8);
    async_b128(lds_offset(&vbuf[buf][0]) + (unsigned)tid * 16,
               vp + (size_t)vrow * S_ + j + vc8 * 8);
  };

  const int ntiles = (q0 + 128) >> 5;  // block-uniform; S%32==0 -> no OOB
  stage(0, 0);
  for (int jt = 0; jt < ntiles; ++jt) {
    const int j = jt * 32;
    const int cur = jt & 1;
    if (jt + 1 < ntiles) {
      stage(cur ^ 1, j + 32);
      asm volatile("s_wait_asynccnt 2" ::: "memory");  // current tile done
    } else {
      asm volatile("s_wait_asynccnt 0" ::: "memory");
    }
    __syncthreads();  // every wave's async loads are now visible

    if (j < qbase + 16) {  // causally active for this wave?
      const half_t* kb = kbuf[cur];
      const half_t* vb = vbuf[cur];

      // ---- S = Q K^T (scaled) ----
      float sv[2][8];
#pragma unroll
      for (int t = 0; t < 2; ++t) {
        v16h b0 = ld_b32x16(kb + t * 16 * 64, 64, nl, g);
        v16h b1 = ld_b32x16(kb + t * 16 * 64 + 32, 64, nl, g);
        v8f s = vzero8();
        s = wmma_f16(qa0, b0, s);
        s = wmma_f16(qa1, b1, s);
#pragma unroll
        for (int i = 0; i < 8; ++i) sv[t][i] = s[i] * 0.125f;  // 1/sqrt(64)
      }
      if (j + 32 > qbase) {  // diagonal tile: causal mask
#pragma unroll
        for (int t = 0; t < 2; ++t)
#pragma unroll
          for (int i = 0; i < 8; ++i)
            if (j + t * 16 + nl > qbase + g * 8 + i) sv[t][i] = -1.0e30f;
      }

      // ---- online softmax ----
#pragma unroll
      for (int i = 0; i < 8; ++i) {
        float rm = rmax16(fmaxf(sv[0][i], sv[1][i]));
        float mn = fmaxf(m_i[i], rm);
        float alpha = __expf(m_i[i] - mn);
        m_i[i] = mn;
        float p0 = __expf(sv[0][i] - mn);
        float p1 = __expf(sv[1][i] - mn);
        sv[0][i] = p0; sv[1][i] = p1;
        l_i[i] = l_i[i] * alpha + rsum16(p0 + p1);
#pragma unroll
        for (int d = 0; d < 4; ++d) acc[d][i] *= alpha;
      }

      // ---- P: C-layout -> A-layout through wave-private LDS ----
#pragma unroll
      for (int t = 0; t < 2; ++t)
#pragma unroll
        for (int i = 0; i < 8; ++i)
          myp[(g * 8 + i) * 32 + t * 16 + nl] = (half_t)sv[t][i];
      asm volatile("s_wait_dscnt 0" ::: "memory");  // same-wave LDS in-order
      v16h pa = ld_a16x32(myp, 32, nl, g);

      // ---- O += P V ----
#pragma unroll
      for (int dt = 0; dt < 4; ++dt) {
        v16h vfr = ld_b32x16(vb + dt * 16 * 32, 32, nl, g);
        acc[dt] = wmma_f16(pa, vfr, acc[dt]);
      }
    }
    __syncthreads();  // all waves done reading buf[cur] before it is reused
  }

  // ---- epilogue: normalize, store heads concatenated [B,S,H*D] ----
  const int b_ = bh / H_;
  const int h = bh % H_;
  half_t* op = attn + ((size_t)b_ * S_ + qbase) * HD_ + h * D_;
#pragma unroll
  for (int i = 0; i < 8; ++i) {
    const float inv = 1.0f / l_i[i];
#pragma unroll
    for (int dt = 0; dt < 4; ++dt)
      op[(size_t)(g * 8 + i) * HD_ + dt * 16 + nl] = (half_t)(acc[dt][i] * inv);
  }
}

// ---------------- output projection ----------------------------------------
// attn [B*S, HD] f16 ; wpT [E, HD] f16 ; out [B*S, E] f32 (+bias).
__global__ __launch_bounds__(256)
void proj_gemm(const half_t* __restrict__ attn, const half_t* __restrict__ wpT,
               const float* __restrict__ bp, float* __restrict__ out) {
  const int lane = threadIdx.x & 31;
  const int wave = threadIdx.x >> 5;
  const int nl = lane & 15, g = lane >> 4;
  const int m0 = blockIdx.x * 128 + wave * 16;
  const int n0 = blockIdx.y * 64;

  v8f acc[4];
#pragma unroll
  for (int n = 0; n < 4; ++n) acc[n] = vzero8();

  const half_t* aa = attn + (size_t)m0 * HD_;
  for (int kc = 0; kc < HD_ / 32; ++kc) {
    v16h a = ld_a16x32(aa + kc * 32, HD_, nl, g);
#pragma unroll
    for (int n = 0; n < 4; ++n) {
      v16h b = ld_b32x16(wpT + (size_t)(n0 + n * 16) * HD_ + kc * 32, HD_, nl, g);
      acc[n] = wmma_f16(a, b, acc[n]);
    }
  }
#pragma unroll
  for (int n = 0; n < 4; ++n) {
    const float bias = bp[n0 + n * 16 + nl];
#pragma unroll
    for (int i = 0; i < 8; ++i)
      out[(size_t)(m0 + g * 8 + i) * E_ + n0 + n * 16 + nl] = acc[n][i] + bias;
  }
}

// ---------------- host launch ----------------------------------------------

extern "C" void kernel_launch(void* const* d_in, const int* in_sizes, int n_in,
                              void* d_out, int out_size, void* d_ws, size_t ws_size,
                              hipStream_t stream) {
  const float* x  = (const float*)d_in[0];
  const float* Wq = (const float*)d_in[1];
  const float* bq = (const float*)d_in[2];
  const float* Wk = (const float*)d_in[3];
  const float* bk = (const float*)d_in[4];
  const float* Wv = (const float*)d_in[5];
  const float* bv = (const float*)d_in[6];
  const float* Wp = (const float*)d_in[7];
  const float* bp = (const float*)d_in[8];
  float* out = (float*)d_out;

  char* ws = (char*)d_ws;
  size_t off = 0;
  auto alloc = [&](size_t bytes) {
    char* p = ws + off;
    off += (bytes + 255) & ~size_t(255);
    return p;
  };
  half_t* xh  = (half_t*)alloc((size_t)B_ * S_ * E_ * 2);   // 16 MB
  half_t* wqT = (half_t*)alloc((size_t)H_ * D_ * E_ * 2);   //  2 MB
  half_t* wkT = (half_t*)alloc((size_t)H_ * D_ * E_ * 2);
  half_t* wvT = (half_t*)alloc((size_t)H_ * D_ * E_ * 2);
  half_t* wpT = (half_t*)alloc((size_t)E_ * HD_ * 2);       //  2 MB
  half_t* qb  = (half_t*)alloc((size_t)B_ * H_ * S_ * D_ * 2);  // 16 MB
  half_t* kb  = (half_t*)alloc((size_t)B_ * H_ * S_ * D_ * 2);
  half_t* vtb = (half_t*)alloc((size_t)B_ * H_ * D_ * S_ * 2);
  half_t* at  = (half_t*)alloc((size_t)B_ * S_ * HD_ * 2);

  // 1) precision conversion + weight transposes
  {
    int n = B_ * S_ * E_;
    cvt_f32_to_f16<<<(n + 255) / 256, 256, 0, stream>>>(x, xh, n);
    int wn = H_ * E_ * D_;
    transpose_w<<<(wn + 255) / 256, 256, 0, stream>>>(Wq, wqT, H_, E_, D_);
    transpose_w<<<(wn + 255) / 256, 256, 0, stream>>>(Wk, wkT, H_, E_, D_);
    transpose_w<<<(wn + 255) / 256, 256, 0, stream>>>(Wv, wvT, H_, E_, D_);
    int pn = HD_ * E_;
    transpose_w<<<(pn + 255) / 256, 256, 0, stream>>>(Wp, wpT, 1, HD_, E_);
  }

  // 2) fused QKV projection
  qkv_gemm<<<dim3((B_ * S_) / 128, H_), 256, 0, stream>>>(
      xh, wqT, wkT, wvT, bq, bk, bv, qb, kb, vtb);

  // 3) flash attention (causal) with async double-buffered K/V staging
  flash_attn<<<dim3(S_ / 128, B_ * H_), 256, 0, stream>>>(qb, kb, vtb, at);

  // 4) output projection + bias
  proj_gemm<<<dim3((B_ * S_) / 128, E_ / 64), 256, 0, stream>>>(at, wpT, bp, out);
}
